// RelationalProposalModule_64665027608821
// MI455X (gfx1250) — matrix operand order repaired and checked by reading, not compiled, not run on hardware
//
#include <hip/hip_runtime.h>
#include <hip/hip_bf16.h>
#include <math.h>

// ---------------------------------------------------------------------------
// RelationalProposalModule on MI455X (gfx1250).
// Memory-bound on the 402MB position_embedding read (~17us at 23.3 TB/s).
// All GEMM phases use v_wmma_f32_16x16x32_bf16 (bf16 in, fp32 accumulate).
// v3: pre-transposed bf16 weights (all GEMM fragments now contiguous b128
//     loads), bf16 LDS probability staging (phase-3 A-frags = 2x ds_load_b128),
//     vectorized epilogues.
// ---------------------------------------------------------------------------

typedef __bf16 bf16_t;
typedef __attribute__((ext_vector_type(16))) __bf16 v16bf;
typedef __attribute__((ext_vector_type(8)))  __bf16 v8bf;
typedef __attribute__((ext_vector_type(8)))  float  v8f;
typedef __attribute__((ext_vector_type(4)))  float  f32x4;

constexpr int Bc = 4, Kc = 512, Dc = 768, DKc = 96, DGc = 96, Hc = 8;

__device__ __forceinline__ v8f wmma_bf16(v16bf a, v16bf b, v8f c) {
  // (neg_a, A, neg_b, B, c_mod, C, reuse_a, reuse_b)
  return __builtin_amdgcn_wmma_f32_16x16x32_bf16(false, a, false, b, (short)0, c,
                                                 false, false);
}

__device__ __forceinline__ v16bf cat8(v8bf lo, v8bf hi) {
  return __builtin_shufflevector(lo, hi, 0, 1, 2, 3, 4, 5, 6, 7,
                                 8, 9, 10, 11, 12, 13, 14, 15);
}

// A-fragment (16-bit A layout) from an fp32 row: elements [off..off+7] and
// [off+16..off+23]; off is a multiple of 8 -> four aligned float4 loads.
__device__ __forceinline__ v16bf afrag_f32(const float* __restrict__ row, int off) {
  const f32x4* p = (const f32x4*)(row + off);
  const f32x4* q = (const f32x4*)(row + off + 16);
  f32x4 x0 = p[0], x1 = p[1], x2 = q[0], x3 = q[1];
  v16bf a;
#pragma unroll
  for (int u = 0; u < 4; ++u) {
    a[u]      = (bf16_t)x0[u];
    a[4 + u]  = (bf16_t)x1[u];
    a[8 + u]  = (bf16_t)x2[u];
    a[12 + u] = (bf16_t)x3[u];
  }
  return a;
}

// 16 consecutive fp32 values -> v16bf.
__device__ __forceinline__ v16bf frag16_f32(const float* __restrict__ p16) {
  const f32x4* p = (const f32x4*)p16;
  f32x4 x0 = p[0], x1 = p[1], x2 = p[2], x3 = p[3];
  v16bf b;
#pragma unroll
  for (int u = 0; u < 4; ++u) {
    b[u]      = (bf16_t)x0[u];
    b[4 + u]  = (bf16_t)x1[u];
    b[8 + u]  = (bf16_t)x2[u];
    b[12 + u] = (bf16_t)x3[u];
  }
  return b;
}

// A-fragment from a bf16 row: [off..off+7] ++ [off+16..off+23], 16B loads.
__device__ __forceinline__ v16bf afrag_bf16(const bf16_t* row, int off) {
  v8bf lo = *(const v8bf*)(row + off);
  v8bf hi = *(const v8bf*)(row + off + 16);
  return cat8(lo, hi);
}

// 16 consecutive bf16 values (32B aligned) -> v16bf.
__device__ __forceinline__ v16bf frag16_bf16(const bf16_t* p16) {
  v8bf lo = *(const v8bf*)p16;
  v8bf hi = *(const v8bf*)(p16 + 8);
  return cat8(lo, hi);
}

// ---------------------------------------------------------------------------
// Kernel 0: one-off weight transpose (3,H,D,DK) fp32 -> (3,H,DK,D) bf16 so
// projection B-fragments are contiguous. Coalesced fp32 reads (lanes span the
// fast DK axis), 16B bf16 vector writes. 221184 threads, ~10.6MB total traffic.
// ---------------------------------------------------------------------------
__global__ __launch_bounds__(256) void wtrans_kernel(
    const float* __restrict__ WK_w, const float* __restrict__ WQ_w,
    const float* __restrict__ WV_w, bf16_t* __restrict__ wt)
{
  int t = blockIdx.x * blockDim.x + threadIdx.x;   // 3*H*(D/8)*DK
  const int f  = t % DKc;     t /= DKc;
  const int dc = t % (Dc/8);  t /= (Dc/8);
  const int h  = t % Hc;
  const int w  = t / Hc;
  const float* Wp  = (w == 0) ? WK_w : (w == 1) ? WQ_w : WV_w;
  const float* src = Wp + ((size_t)h * Dc + dc * 8) * DKc + f;
  v8bf v;
#pragma unroll
  for (int u = 0; u < 8; ++u) v[u] = (bf16_t)src[(size_t)u * DKc];
  *(v8bf*)(wt + (((size_t)w * Hc + h) * DKc + f) * Dc + dc * 8) = v;
}

// ---------------------------------------------------------------------------
// Kernel 1: per-head K/Q/V projections.
//   (B,K,D) x (H,D,DK) -> bf16 (B,H,K,DK); V stored transposed (B,H,DK,K).
// One wave per 16x16 output tile; 24 K-chunks of 32 over D=768. A-frags are
// 4x b128 fp32 loads + cvt; B-frags are 2x b128 bf16 loads from wt.
// ---------------------------------------------------------------------------
__global__ __launch_bounds__(256) void proj_kernel(
    const float* __restrict__ f_a, const bf16_t* __restrict__ wt,
    const float* __restrict__ WK_b, const float* __restrict__ WQ_b,
    const float* __restrict__ WV_b,
    bf16_t* __restrict__ wk, bf16_t* __restrict__ wq, bf16_t* __restrict__ wvT)
{
  const int lane = threadIdx.x & 31;
  int gid = blockIdx.x * (blockDim.x >> 5) + (threadIdx.x >> 5);
  const int nt = gid % 6;  gid /= 6;        // DK/16 tiles
  const int mt = gid % 32; gid /= 32;       // K/16 tiles
  const int h  = gid % 8;  gid /= 8;
  const int b  = gid % 4;
  const int w  = gid / 4;                   // 0:K 1:Q 2:V

  const float* bp = (w == 0) ? WK_b : (w == 1) ? WQ_b : WV_b;

  const int jn     = lane & 15;
  const int hihalf = (lane & 16) ? 1 : 0;
  const int abase  = hihalf ? 16 : 0;       // A-frag K base (contiguous)
  const int bbase  = hihalf ? 16 : 0;       // B-frag K base
  const int rbase  = hihalf ? 8 : 0;        // C/D row base

  const float* xrow = f_a + ((size_t)b * Kc + mt * 16 + jn) * Dc;   // A row
  const int    col  = nt * 16 + jn;                                  // B/D col
  const bf16_t* wtrow = wt + (((size_t)w * Hc + h) * DKc + col) * Dc;

  v8f acc = {};
  for (int kk = 0; kk < Dc / 32; ++kk) {
    const int kb = kk * 32;
    acc = wmma_bf16(frag16_f32(xrow + kb + abase),
                    frag16_bf16(wtrow + kb + bbase), acc);
  }

  const float bias = bp[h * DKc + col];
  if (w < 2) {
    bf16_t* dst = (w == 0) ? wk : wq;
#pragma unroll
    for (int r = 0; r < 8; ++r) {
      const int m = mt * 16 + rbase + r;
      dst[(((size_t)b * Hc + h) * Kc + m) * DKc + col] = (bf16_t)(acc[r] + bias);
    }
  } else {
    v8bf pk;
#pragma unroll
    for (int r = 0; r < 8; ++r) pk[r] = (bf16_t)(acc[r] + bias);
    *(v8bf*)(wvT + (((size_t)b * Hc + h) * DKc + col) * Kc + mt * 16 + rbase) = pk;
  }
}

// ---------------------------------------------------------------------------
// Kernel 2: geometric gate, all 8 heads in ONE pass over position_embedding.
//   gate[b,h,i,j] = log(max(pe[b,i,j,:].WG_w[h] + WG_b[h], 1e-6))
// Tall-skinny GEMM (B*K*K, 96) x (96, 8 padded to 16) via WMMA; pe rows are
// contiguous 96-float runs -> b128 streaming of the 402MB tensor.
// ---------------------------------------------------------------------------
__global__ __launch_bounds__(256) void gate_kernel(
    const float* __restrict__ pe, const float* __restrict__ WG_w,
    const float* __restrict__ WG_b, float* __restrict__ gate)
{
  const int lane  = threadIdx.x & 31;
  const int gtile = blockIdx.x * (blockDim.x >> 5) + (threadIdx.x >> 5);
  const size_t p0 = (size_t)gtile * 16;            // 16 (i,j) pairs per tile
  const int    b  = (int)(p0 / ((size_t)Kc * Kc));
  const size_t pr = p0 - (size_t)b * Kc * Kc;

  const int jn     = lane & 15;
  const int hihalf = (lane & 16) ? 1 : 0;
  const int abase  = hihalf ? 16 : 0;
  const int bbase  = hihalf ? 16 : 0;
  const int rbase  = hihalf ? 8 : 0;

  const float* perow = pe + (p0 + jn) * DGc;       // contiguous 96 floats

  v8f acc = {};
#pragma unroll
  for (int kk = 0; kk < 3; ++kk) {
    const int kb = kk * 32;
    v16bf a = frag16_f32(perow + kb + abase);
    v16bf bb;
    if (jn < 8) {
      bb = frag16_f32(WG_w + jn * DGc + kb + bbase);
    } else {
#pragma unroll
      for (int u = 0; u < 16; ++u) bb[u] = (bf16_t)0.0f;
    }
    acc = wmma_bf16(a, bb, acc);
  }

  if (jn < 8) {                                    // columns 8..15 are padding
    const float gb = WG_b[jn];
    float* dst = gate + ((size_t)b * Hc + jn) * Kc * Kc + pr + rbase;
    f32x4 o0, o1;
#pragma unroll
    for (int r = 0; r < 4; ++r) {
      o0[r] = __logf(fmaxf(acc[r] + gb, 1e-6f));
      o1[r] = __logf(fmaxf(acc[4 + r] + gb, 1e-6f));
    }
    *(f32x4*)dst       = o0;                       // 32B-aligned runs
    *(f32x4*)(dst + 4) = o1;
  }
}

// ---------------------------------------------------------------------------
// Kernel 3: gated attention, one workgroup (4 waves) per (b, h, j-tile).
// Scores S[i,j] = <wk[i], wq[j]>/sqrt(96) + gate[i,j] built in fp32 LDS
// (stride 20 avoids lane-half bank conflicts), softmax over i=512, then the
// normalize pass packs probs to bf16 LDS in column-major fragment order so
// phase-3 A-frags are 2x ds_load_b128; each A-frag feeds two f-tiles.
// ---------------------------------------------------------------------------
#define LDSTR 20
#define PBSTR 520   // 16 bf16 columns, padded: 520*2B rows fan across banks
__global__ __launch_bounds__(128) void attn_kernel(
    const bf16_t* __restrict__ wk, const bf16_t* __restrict__ wq,
    const bf16_t* __restrict__ wvT, const float* __restrict__ gate,
    const float* __restrict__ f_a, const float* __restrict__ gammap,
    float* __restrict__ out)
{
  __shared__ float P[Kc * LDSTR];                 // fp32 logits (i-major)
  __shared__ __align__(16) bf16_t Pbf[16 * PBSTR]; // bf16 probs (j-major)
  __shared__ float pred[2][8 * 16];               // partial max / partial sum
  __shared__ float cmax[16];
  __shared__ float cinv[16];

  const int tid  = threadIdx.x;
  const int lane = tid & 31;
  const int wave = tid >> 5;
  const int jt = blockIdx.x & 31;
  const int h  = (blockIdx.x >> 5) & 7;
  const int b  = blockIdx.x >> 8;
  const size_t bh = (size_t)b * Hc + h;

  const int jn     = lane & 15;
  const int hihalf = (lane & 16) ? 1 : 0;
  const int abase  = hihalf ? 16 : 0;
  const int bbase  = hihalf ? 16 : 0;
  const int rbase  = hihalf ? 8 : 0;
  const float scale = 0.10206207261596575f;        // 1/sqrt(96)

  // Loop-invariant wq B-fragments (3 K-chunks), loaded once.
  const bf16_t* wqrow = wq + (bh * Kc + jt * 16 + jn) * DKc;
  v16bf bq[3];
#pragma unroll
  for (int kk = 0; kk < 3; ++kk)
    bq[kk] = frag16_bf16(wqrow + kk * 32 + bbase);

  // ---- phase 1: score tiles (each wave owns 8 of the 32 i-tiles) ----
  for (int it = wave; it < 32; it += 4) {
    const bf16_t* wkrow = wk + (bh * Kc + it * 16 + jn) * DKc;
    v8f acc = {};
#pragma unroll
    for (int kk = 0; kk < 3; ++kk)
      acc = wmma_bf16(frag16_bf16(wkrow + kk * 32 + abase), bq[kk], acc);

    const float* grow = gate + bh * Kc * Kc +
                        (size_t)(it * 16 + rbase) * Kc + jt * 16 + jn;
#pragma unroll
    for (int r = 0; r < 8; ++r) {
      const int i = it * 16 + rbase + r;
      P[i * LDSTR + jn] = acc[r] * scale + grow[(size_t)r * Kc];
    }
  }
  __syncthreads();

  // ---- phase 2: softmax over i (128 threads = 16 cols x 8 strips) ----
  {
    const int j = tid & 15, s = tid >> 4;
    float m = -1e30f;
    for (int ii = 0; ii < 64; ++ii)
      m = fmaxf(m, P[(s * 64 + ii) * LDSTR + j]);
    pred[0][s * 16 + j] = m;
  }
  __syncthreads();
  if (tid < 16) {
    float m = pred[0][tid];
    for (int s = 1; s < 8; ++s) m = fmaxf(m, pred[0][s * 16 + tid]);
    cmax[tid] = m;
  }
  __syncthreads();
  {
    const int j = tid & 15, s = tid >> 4;
    const float m = cmax[j];
    float sum = 0.f;
    for (int ii = 0; ii < 64; ++ii) {
      const int idx = (s * 64 + ii) * LDSTR + j;
      const float e = __expf(P[idx] - m);
      P[idx] = e;
      sum += e;
    }
    pred[1][s * 16 + j] = sum;
  }
  __syncthreads();
  if (tid < 16) {
    float sum = 0.f;
    for (int s = 0; s < 8; ++s) sum += pred[1][s * 16 + tid];
    cinv[tid] = 1.0f / sum;
  }
  __syncthreads();
  {  // normalize + pack to bf16, column-major: Pbf[j][i] = P[i][j] * inv[j]
    const int j = tid & 15, s = tid >> 4;
    const float inv = cinv[j];
    bf16_t* prow = Pbf + j * PBSTR + s * 64;       // 16B aligned
    for (int ii = 0; ii < 64; ii += 8) {
      v8bf v;
#pragma unroll
      for (int u = 0; u < 8; ++u)
        v[u] = (bf16_t)(P[(s * 64 + ii + u) * LDSTR + j] * inv);
      *(v8bf*)(prow + ii) = v;
    }
  }
  __syncthreads();

  const float gamma = gammap[0];

  // ---- phase 3: out[j,f] = sum_i P[i,j]*v[i,f]; each wave owns f-tiles
  //      {wave, wave+4}; the A-fragment (2x ds_load_b128) feeds both tiles. --
  const int  nt0 = wave;
  const bool two = (wave + 4) < 6;                 // wave-uniform
  const int  nt1 = two ? wave + 4 : wave;
  const bf16_t* vrow0 = wvT + (bh * DKc + nt0 * 16 + jn) * Kc;
  const bf16_t* vrow1 = wvT + (bh * DKc + nt1 * 16 + jn) * Kc;
  const bf16_t* prow  = Pbf + jn * PBSTR;
  v8f acc0 = {}, acc1 = {};
  for (int kc = 0; kc < 16; ++kc) {                // K-dim = i = 512
    const int kb = kc * 32;
    v16bf a = frag16_bf16(prow + kb + abase);
    acc0 = wmma_bf16(a, frag16_bf16(vrow0 + kb + bbase), acc0);
    if (two)
      acc1 = wmma_bf16(a, frag16_bf16(vrow1 + kb + bbase), acc1);
  }
#pragma unroll
  for (int r = 0; r < 8; ++r) {
    const int j = jt * 16 + rbase + r;
    const size_t o = ((size_t)b * Kc + j) * Dc + h * DKc + nt0 * 16 + jn;
    out[o] = gamma * acc0[r] + f_a[o];             // residual epilogue
  }
  if (two) {
#pragma unroll
    for (int r = 0; r < 8; ++r) {
      const int j = jt * 16 + rbase + r;
      const size_t o = ((size_t)b * Kc + j) * Dc + h * DKc + nt1 * 16 + jn;
      out[o] = gamma * acc1[r] + f_a[o];
    }
  }
}

// ---------------------------------------------------------------------------
extern "C" void kernel_launch(void* const* d_in, const int* in_sizes, int n_in,
                              void* d_out, int out_size, void* d_ws, size_t ws_size,
                              hipStream_t stream) {
  const float* f_a   = (const float*)d_in[0];
  const float* pe    = (const float*)d_in[1];
  const float* WG_w  = (const float*)d_in[2];
  const float* WG_b  = (const float*)d_in[3];
  const float* WK_w  = (const float*)d_in[4];
  const float* WK_b  = (const float*)d_in[5];
  const float* WQ_w  = (const float*)d_in[6];
  const float* WQ_b  = (const float*)d_in[7];
  const float* WV_w  = (const float*)d_in[8];
  const float* WV_b  = (const float*)d_in[9];
  const float* gamma = (const float*)d_in[10];
  float* out = (float*)d_out;

  // Workspace layout (~47 MB total):
  //   [0)      wk  : B*H*K*DK bf16      (3 MB)
  //   [3MB)    wq  : B*H*K*DK bf16      (3 MB)
  //   [6MB)    wvT : B*H*DK*K bf16      (3 MB, transposed)
  //   [9MB)    gate: B*H*K*K  fp32      (33.5 MB)
  //   [42.5MB) wt  : 3*H*DK*D bf16      (3.5 MB, transposed weights)
  char* ws = (char*)d_ws;
  const size_t nproj = (size_t)Bc * Hc * Kc * DKc;
  const size_t ngate = (size_t)Bc * Hc * Kc * Kc;
  bf16_t* wk   = (bf16_t*)ws;
  bf16_t* wq   = (bf16_t*)(ws + nproj * sizeof(bf16_t));
  bf16_t* wvT  = (bf16_t*)(ws + 2 * nproj * sizeof(bf16_t));
  float*  gate = (float*)(ws + 3 * nproj * sizeof(bf16_t));
  bf16_t* wt   = (bf16_t*)(ws + 3 * nproj * sizeof(bf16_t) + ngate * sizeof(float));

  // 3*H*(D/8)*DK = 221184 threads
  wtrans_kernel<<<221184 / 256, 256, 0, stream>>>(WK_w, WQ_w, WV_w, wt);
  // 3 (K/Q/V) * B * H * (K/16) * (DK/16) = 18432 tiles, 8 waves/block
  proj_kernel<<<18432 / 8, 256, 0, stream>>>(f_a, wt, WK_b, WQ_b, WV_b,
                                             wk, wq, wvT);
  // B*K*K/16 = 65536 pair-tiles, 8 waves/block
  gate_kernel<<<65536 / 8, 256, 0, stream>>>(pe, WG_w, WG_b, gate);
  // B * H * (K/16) = 1024 workgroups of 4 waves
  attn_kernel<<<1024, 128, 0, stream>>>(wk, wq, wvT, gate, f_a, gamma, out);
}
